// SmoothingProposal_65627100283107
// MI455X (gfx1250) — matrix operand order repaired and checked by reading, not compiled, not run on hardware
//
#include <hip/hip_runtime.h>

// ---------------- problem constants (from reference) ----------------
#define BB   128
#define TT   512
#define OBS  64
#define HH   256
#define DD   16
#define KK   3
#define PHH  64

typedef __attribute__((ext_vector_type(16))) __bf16 v16bf;
typedef __attribute__((ext_vector_type(8)))  float  v8f;

union FragAB { uint4 u[2]; v16bf v; unsigned short s[16]; };
union FragC  { v8f v; float f[8]; };

__device__ __forceinline__ unsigned short f2bf(float f) {
    union { float f; unsigned u; } x; x.f = f;
    unsigned u = x.u;
    return (unsigned short)((u + 0x7FFFu + ((u >> 16) & 1u)) >> 16);
}
__device__ __forceinline__ float bf2f(unsigned short s) {
    union { unsigned u; float f; } x; x.u = ((unsigned)s) << 16; return x.f;
}
__device__ __forceinline__ float sigmoidf(float x) { return 1.f / (1.f + __expf(-x)); }
__device__ __forceinline__ float softplusf(float x) { return (x > 20.f) ? x : log1pf(__expf(x)); }

__device__ __forceinline__ v8f wmma_bf16(v16bf a, v16bf b, v8f c) {
    return __builtin_amdgcn_wmma_f32_16x16x32_bf16(false, a, false, b, (short)0, c, false, false);
}

// compiler-level memory barrier: keeps weight-fragment loads inside the time
// loop (prevents LICM hoist + scratch spill of loop-invariant loads)
#define LOOP_FENCE() asm volatile("" ::: "memory")

// ---- B fragment load from packed layout:
// packed[ tile*(Kp*16) + kk*512 + lane*16 + j ] = B[k = kk*32 + (lane/16)*16 + j][n = tile*16 + lane%16]
__device__ __forceinline__ v16bf load_bfrag(const unsigned short* __restrict__ p,
                                            int tile, int Kp, int kk, int lane) {
    const uint4* q = (const uint4*)(p + (size_t)tile * Kp * 16 + kk * 512 + lane * 16);
    FragAB f; f.u[0] = q[0]; f.u[1] = q[1];
    return f.v;
}

// ---- A fragment from row-major bf16 [rows][stride]; rows = M (lane%16)
__device__ __forceinline__ v16bf load_afrag_bf16(const unsigned short* base, int stride,
                                                 int kk, int lane) {
    int m  = lane & 15;
    int kb = kk * 32 + ((lane & 16) ? 16 : 0);
    const unsigned short* r = base + (size_t)m * stride + kb;
    FragAB f;
    f.u[0] = *(const uint4*)(r);
    f.u[1] = *(const uint4*)(r + 8);
    return f.v;
}

// ---- A fragment from row-major f32 (observations), converted to bf16
__device__ __forceinline__ v16bf load_afrag_f32(const float* base, int stride,
                                                int kk, int lane) {
    int m  = lane & 15;
    int kb = kk * 32 + ((lane & 16) ? 16 : 0);
    const float4* p0 = (const float4*)(base + (size_t)m * stride + kb);
    const float4* p1 = (const float4*)(base + (size_t)m * stride + kb + 8);
    float4 a0 = p0[0], a1 = p0[1], b0 = p1[0], b1 = p1[1];
    FragAB f;
    f.s[0] = f2bf(a0.x); f.s[1] = f2bf(a0.y); f.s[2] = f2bf(a0.z); f.s[3] = f2bf(a0.w);
    f.s[4] = f2bf(a1.x); f.s[5] = f2bf(a1.y); f.s[6] = f2bf(a1.z); f.s[7] = f2bf(a1.w);
    f.s[8]  = f2bf(b0.x); f.s[9]  = f2bf(b0.y); f.s[10] = f2bf(b0.z); f.s[11] = f2bf(b0.w);
    f.s[12] = f2bf(b1.x); f.s[13] = f2bf(b1.y); f.s[14] = f2bf(b1.z); f.s[15] = f2bf(b1.w);
    return f.v;
}

// ---- A fragment for K=16 state (padded to K=32 with zeros), from LDS [16][16]
__device__ __forceinline__ v16bf load_afrag_state(const unsigned short* s, int lane) {
    int m  = lane & 15;
    FragAB f;
    if (lane & 16) { f.u[0] = make_uint4(0u, 0u, 0u, 0u); f.u[1] = make_uint4(0u, 0u, 0u, 0u); }
    else { f.u[0] = *(const uint4*)(s + m * 16); f.u[1] = *(const uint4*)(s + m * 16 + 8); }
    return f.v;
}

// ================= weight pack kernel =================
// dst index: (((n0*(Kp/32) + kk)*2 + h)*16 + nn)*16 + j ; k = kk*32+h*16+j ; n = n0*16+nn
__global__ void pack_b_kernel(const float* __restrict__ src, unsigned short* __restrict__ dst,
                              int realK, int realN, int Kp, int Np, int transposed, float scale) {
    int e = blockIdx.x * blockDim.x + threadIdx.x;
    int total = Kp * Np;
    if (e >= total) return;
    int j    = e & 15;
    int nn   = (e >> 4) & 15;
    int h    = (e >> 8) & 1;
    int rest = e >> 9;
    int kkc  = Kp >> 5;
    int kk   = rest % kkc;
    int n0   = rest / kkc;
    int k = kk * 32 + h * 16 + j;
    int n = n0 * 16 + nn;
    float v = 0.f;
    if (k < realK && n < realN)
        v = (transposed ? src[(size_t)n * realK + k] : src[(size_t)k * realN + n]) * scale;
    dst[e] = f2bf(v);
}

// ================= LSTM layer scan =================
// grid = 8 workgroups (16 batch rows each), block = 512 (16 waves).
// Wave w owns h columns [16w, 16w+16); computes 4 gate tiles (i,f,g,o) per step.
__global__ __launch_bounds__(512) void lstm_scan_kernel(
    const float* __restrict__ xf32,            // layer0: observations [B][T][OBS], else null
    const unsigned short* __restrict__ xbf,    // layer1: h1s [T][B][H] bf16, else null
    const unsigned short* __restrict__ Pih, int KpIh,
    const unsigned short* __restrict__ Phh,
    const float* __restrict__ bias,            // [4H]
    const float* __restrict__ h0,              // [H] (already layer-offset)
    const float* __restrict__ c0,              // [H]
    unsigned short* __restrict__ hout,         // [T][B][H] bf16
    int inF32)
{
    __shared__ __align__(16) unsigned short hshm[16 * HH];   // 8 KB
    const int tid  = threadIdx.x;
    const int lane = tid & 31;
    const int w    = tid >> 5;
    const int wg   = blockIdx.x;
    const int col  = w * 16 + (lane & 15);

    for (int e = tid; e < 16 * HH; e += 512) hshm[e] = f2bf(h0[e & (HH - 1)]);

    float bval[4];
#pragma unroll
    for (int g = 0; g < 4; ++g) bval[g] = bias[g * HH + col];

    FragC cst;
    {
        float c0v = c0[col];
#pragma unroll
        for (int r = 0; r < 8; ++r) cst.f[r] = c0v;
    }
    __syncthreads();

    for (int t = 0; t < TT; ++t) {
        LOOP_FENCE();   // keep weight loads in-loop (global/L2), not hoisted+spilled

        FragC acc[4];
#pragma unroll
        for (int g = 0; g < 4; ++g)
#pragma unroll
            for (int r = 0; r < 8; ++r) acc[g].f[r] = 0.f;

        // ---- recurrent path: h_{t-1} @ Whh^T (K = 256), B frags pipelined 1 ahead
        {
            v16bf bcur[4], bnxt[4];
#pragma unroll
            for (int g = 0; g < 4; ++g) bcur[g] = load_bfrag(Phh, g * 16 + w, HH, 0, lane);
#pragma unroll
            for (int kk = 0; kk < HH / 32; ++kk) {
                v16bf a = load_afrag_bf16(hshm, HH, kk, lane);
                if (kk + 1 < HH / 32) {
#pragma unroll
                    for (int g = 0; g < 4; ++g)
                        bnxt[g] = load_bfrag(Phh, g * 16 + w, HH, kk + 1, lane);
                }
#pragma unroll
                for (int g = 0; g < 4; ++g)
                    acc[g].v = wmma_bf16(a, bcur[g], acc[g].v);
#pragma unroll
                for (int g = 0; g < 4; ++g) bcur[g] = bnxt[g];
            }
        }
        // ---- input path
        if (inF32) {
            const float* xb = xf32 + (size_t)(wg * 16) * TT * OBS + (size_t)t * OBS;
            if (t + 1 < TT) __builtin_prefetch(xb + OBS, 0, 0);
#pragma unroll
            for (int kk = 0; kk < OBS / 32; ++kk) {
                v16bf a = load_afrag_f32(xb, TT * OBS, kk, lane);
#pragma unroll
                for (int g = 0; g < 4; ++g) {
                    v16bf b = load_bfrag(Pih, g * 16 + w, KpIh, kk, lane);
                    acc[g].v = wmma_bf16(a, b, acc[g].v);
                }
            }
        } else {
            const unsigned short* xb = xbf + ((size_t)t * BB + wg * 16) * HH;
            if (t + 1 < TT) __builtin_prefetch(xb + (size_t)BB * HH, 0, 0);
            v16bf bcur[4], bnxt[4];
#pragma unroll
            for (int g = 0; g < 4; ++g) bcur[g] = load_bfrag(Pih, g * 16 + w, KpIh, 0, lane);
#pragma unroll
            for (int kk = 0; kk < HH / 32; ++kk) {
                v16bf a = load_afrag_bf16(xb, HH, kk, lane);
                if (kk + 1 < HH / 32) {
#pragma unroll
                    for (int g = 0; g < 4; ++g)
                        bnxt[g] = load_bfrag(Pih, g * 16 + w, KpIh, kk + 1, lane);
                }
#pragma unroll
                for (int g = 0; g < 4; ++g)
                    acc[g].v = wmma_bf16(a, bcur[g], acc[g].v);
#pragma unroll
                for (int g = 0; g < 4; ++g) bcur[g] = bnxt[g];
            }
        }
        __syncthreads();   // everyone done reading hshm

#pragma unroll
        for (int r = 0; r < 8; ++r) {
            float iv = acc[0].f[r] + bval[0];
            float fv = acc[1].f[r] + bval[1];
            float gv = acc[2].f[r] + bval[2];
            float ov = acc[3].f[r] + bval[3];
            float c  = sigmoidf(fv) * cst.f[r] + sigmoidf(iv) * tanhf(gv);
            float h  = sigmoidf(ov) * tanhf(c);
            cst.f[r] = c;
            int m = r + ((lane & 16) ? 8 : 0);
            unsigned short hb = f2bf(h);
            hshm[m * HH + col] = hb;
            hout[((size_t)t * BB + wg * 16 + m) * HH + col] = hb;
        }
        __syncthreads();
    }
}

// ================= proposal scan =================
// grid = 8 workgroups (16 rows), block = 256 (8 waves)
__global__ __launch_bounds__(256) void proposal_scan_kernel(
    const unsigned short* __restrict__ hf,     // [T][B][H] bf16
    const float* __restrict__ eps0,            // [B][D]
    const float* __restrict__ eps_seq,         // [T][B][D]
    const float* __restrict__ gumbel,          // [T][B][K]
    const unsigned short* __restrict__ Pse,    // Kp=32 (K=16 pad), N=256
    const unsigned short* __restrict__ Ph1,    // Kp=256, N=64  (x0.5 folded)
    const unsigned short* __restrict__ Ph2,    // Kp=64,  N=64
    const unsigned short* __restrict__ Plog,   // Kp=64,  N=16 (pad)
    const unsigned short* __restrict__ Pmu,    // Kp=64,  N=48
    const unsigned short* __restrict__ Psig,   // Kp=64,  N=48
    const unsigned short* __restrict__ Ppm,    // Kp=256, N=16 (x0.5 folded)
    const unsigned short* __restrict__ Pps,    // Kp=256, N=16 (x0.5 folded)
    const float* __restrict__ bse, const float* __restrict__ bh1, const float* __restrict__ bh2,
    const float* __restrict__ blog, const float* __restrict__ bmu, const float* __restrict__ bsig,
    const float* __restrict__ bpm, const float* __restrict__ bps,
    float* __restrict__ out)
{
    __shared__ __align__(16) unsigned short sS[16 * 16];      // current state, bf16
    __shared__ __align__(16) unsigned short sEnc[16 * HH];    // enc + hf, bf16 (8 KB)
    __shared__ __align__(16) unsigned short sH1[16 * PHH];
    __shared__ __align__(16) unsigned short sH2[16 * PHH];
    __shared__ __align__(16) float sLog[16 * 16];
    __shared__ __align__(16) float sMu[16 * 48];
    __shared__ __align__(16) float sSig[16 * 48];
    __shared__ __align__(16) float sTmp[2][16 * 16];

    const int tid  = threadIdx.x;
    const int lane = tid & 31;
    const int w    = tid >> 5;
    const int wg   = blockIdx.x;

    // ---------- s0 = x0@Wpm + bpm + softplus(x0@Wps + bps)*eps0, x0 = 0.5*hf[0] ----------
    if (w < 2) {
        const unsigned short* hfb = hf + (size_t)(wg * 16) * HH;   // t = 0
        const unsigned short* P   = (w == 0) ? Ppm : Pps;
        FragC acc;
#pragma unroll
        for (int r = 0; r < 8; ++r) acc.f[r] = 0.f;
#pragma unroll
        for (int kk = 0; kk < HH / 32; ++kk) {
            v16bf a = load_afrag_bf16(hfb, HH, kk, lane);
            v16bf b = load_bfrag(P, 0, HH, kk, lane);
            acc.v = wmma_bf16(a, b, acc.v);
        }
        const float* bb = (w == 0) ? bpm : bps;
        int n = lane & 15;
        float bv = bb[n];
#pragma unroll
        for (int r = 0; r < 8; ++r) {
            int m = r + ((lane & 16) ? 8 : 0);
            sTmp[w][m * 16 + n] = acc.f[r] + bv;
        }
    }
    __syncthreads();
    {
        int m = tid >> 4, d = tid & 15, b = wg * 16 + m;
        float s = sTmp[0][tid] + softplusf(sTmp[1][tid]) * eps0[b * DD + d];
        out[(size_t)b * (TT + 1) * DD + d] = s;
        sS[tid] = f2bf(s);
    }
    __syncthreads();

    for (int t = 0; t < TT; ++t) {
        LOOP_FENCE();   // keep head/MLP weight loads in-loop, not hoisted+spilled

        // ---- enc = tanh(s@Wse + bse);  store (enc + hf_t) for infos (0.5 folded into Ph1)
        for (int n0 = w; n0 < 16; n0 += 8) {
            FragC acc;
#pragma unroll
            for (int r = 0; r < 8; ++r) acc.f[r] = 0.f;
            v16bf a = load_afrag_state(sS, lane);
            v16bf b = load_bfrag(Pse, n0, 32, 0, lane);
            acc.v = wmma_bf16(a, b, acc.v);
            int n = lane & 15, gn = n0 * 16 + n;
            float bv = bse[gn];
#pragma unroll
            for (int r = 0; r < 8; ++r) {
                int m = r + ((lane & 16) ? 8 : 0);
                float hfv = bf2f(hf[((size_t)t * BB + wg * 16 + m) * HH + gn]);
                sEnc[m * HH + gn] = f2bf(tanhf(acc.f[r] + bv) + hfv);
            }
        }
        __syncthreads();
        // ---- h1 = tanh(infos@Wh1 + bh1)   (K=256), pipelined
        if (w < 4) {
            FragC acc;
#pragma unroll
            for (int r = 0; r < 8; ++r) acc.f[r] = 0.f;
            v16bf bcur = load_bfrag(Ph1, w, HH, 0, lane), bnxt;
#pragma unroll
            for (int kk = 0; kk < HH / 32; ++kk) {
                v16bf a = load_afrag_bf16(sEnc, HH, kk, lane);
                if (kk + 1 < HH / 32) bnxt = load_bfrag(Ph1, w, HH, kk + 1, lane);
                acc.v = wmma_bf16(a, bcur, acc.v);
                bcur = bnxt;
            }
            int n = lane & 15, gn = w * 16 + n;
            float bv = bh1[gn];
#pragma unroll
            for (int r = 0; r < 8; ++r) {
                int m = r + ((lane & 16) ? 8 : 0);
                sH1[m * PHH + gn] = f2bf(tanhf(acc.f[r] + bv));
            }
        }
        __syncthreads();
        // ---- h2 = tanh(h1@Wh2 + bh2)   (K=64)
        if (w < 4) {
            FragC acc;
#pragma unroll
            for (int r = 0; r < 8; ++r) acc.f[r] = 0.f;
#pragma unroll
            for (int kk = 0; kk < PHH / 32; ++kk) {
                v16bf a = load_afrag_bf16(sH1, PHH, kk, lane);
                v16bf b = load_bfrag(Ph2, w, PHH, kk, lane);
                acc.v = wmma_bf16(a, b, acc.v);
            }
            int n = lane & 15, gn = w * 16 + n;
            float bv = bh2[gn];
#pragma unroll
            for (int r = 0; r < 8; ++r) {
                int m = r + ((lane & 16) ? 8 : 0);
                sH2[m * PHH + gn] = f2bf(tanhf(acc.f[r] + bv));
            }
        }
        __syncthreads();
        // ---- heads: logits (wave 0), mu (waves 1..3), sig (waves 4..6)
        if (w < 7) {
            const unsigned short* P;
            int tile; float* dstf; int dstride;
            const float* bb; int bn;
            if (w == 0)      { P = Plog; tile = 0;     dstf = sLog; dstride = 16; bb = blog; bn = KK; }
            else if (w < 4)  { P = Pmu;  tile = w - 1; dstf = sMu;  dstride = 48; bb = bmu;  bn = 48; }
            else             { P = Psig; tile = w - 4; dstf = sSig; dstride = 48; bb = bsig; bn = 48; }
            FragC acc;
#pragma unroll
            for (int r = 0; r < 8; ++r) acc.f[r] = 0.f;
#pragma unroll
            for (int kk = 0; kk < PHH / 32; ++kk) {
                v16bf a = load_afrag_bf16(sH2, PHH, kk, lane);
                v16bf b = load_bfrag(P, tile, PHH, kk, lane);
                acc.v = wmma_bf16(a, b, acc.v);
            }
            int n = lane & 15, gn = tile * 16 + n;
            float bv = (gn < bn) ? bb[gn] : 0.f;
#pragma unroll
            for (int r = 0; r < 8; ++r) {
                int m = r + ((lane & 16) ? 8 : 0);
                dstf[m * dstride + gn] = acc.f[r] + bv;
            }
        }
        __syncthreads();
        // ---- gumbel argmax + reparameterized sample
        {
            int m = tid >> 4, d = tid & 15, b = wg * 16 + m;
            const float* gp = gumbel + ((size_t)t * BB + b) * KK;
            float l0 = sLog[m * 16 + 0] + gp[0];
            float l1 = sLog[m * 16 + 1] + gp[1];
            float l2 = sLog[m * 16 + 2] + gp[2];
            int k = 0; float best = l0;
            if (l1 > best) { best = l1; k = 1; }
            if (l2 > best) { best = l2; k = 2; }
            float muv = sMu[m * 48 + k * DD + d];
            float sgv = sSig[m * 48 + k * DD + d];
            float s = muv + softplusf(sgv) * eps_seq[((size_t)t * BB + b) * DD + d];
            out[((size_t)b * (TT + 1) + (t + 1)) * DD + d] = s;
            sS[tid] = f2bf(s);
        }
        __syncthreads();
    }
}

// ================= host launcher =================
static inline size_t align256(size_t x) { return (x + 255u) & ~(size_t)255u; }

extern "C" void kernel_launch(void* const* d_in, const int* in_sizes, int n_in,
                              void* d_out, int out_size, void* d_ws, size_t ws_size,
                              hipStream_t stream) {
    (void)in_sizes; (void)n_in; (void)out_size; (void)ws_size;
    const float* obs    = (const float*)d_in[0];
    const float* eps0   = (const float*)d_in[1];
    const float* epsS   = (const float*)d_in[2];
    const float* gumb   = (const float*)d_in[3];
    const float* h0     = (const float*)d_in[4];
    const float* c0     = (const float*)d_in[5];
    const float* Wih0   = (const float*)d_in[6];
    const float* Whh0   = (const float*)d_in[7];
    const float* b0     = (const float*)d_in[8];
    const float* Wih1   = (const float*)d_in[9];
    const float* Whh1   = (const float*)d_in[10];
    const float* b1     = (const float*)d_in[11];
    const float* W_se   = (const float*)d_in[12];
    const float* b_se   = (const float*)d_in[13];
    const float* W_pm   = (const float*)d_in[14];
    const float* b_pm   = (const float*)d_in[15];
    const float* W_ps   = (const float*)d_in[16];
    const float* b_ps   = (const float*)d_in[17];
    const float* Wh1    = (const float*)d_in[18];
    const float* bh1    = (const float*)d_in[19];
    const float* Wh2    = (const float*)d_in[20];
    const float* bh2    = (const float*)d_in[21];
    const float* Wlog   = (const float*)d_in[22];
    const float* blog   = (const float*)d_in[23];
    const float* Wmu    = (const float*)d_in[24];
    const float* bmu    = (const float*)d_in[25];
    const float* Wsig   = (const float*)d_in[26];
    const float* bsig   = (const float*)d_in[27];
    float* out = (float*)d_out;

    // ---- workspace carve-up ----
    char* ws = (char*)d_ws;
    size_t off = 0;
    auto alloc = [&](size_t bytes) { char* p = ws + off; off = align256(off + bytes); return p; };
    unsigned short* h1s = (unsigned short*)alloc((size_t)TT * BB * HH * 2);   // layer0 output
    unsigned short* hfb = (unsigned short*)alloc((size_t)TT * BB * HH * 2);   // layer1 output
    unsigned short* P_ih0 = (unsigned short*)alloc((size_t)OBS * 4 * HH * 2);
    unsigned short* P_hh0 = (unsigned short*)alloc((size_t)HH * 4 * HH * 2);
    unsigned short* P_ih1 = (unsigned short*)alloc((size_t)HH * 4 * HH * 2);
    unsigned short* P_hh1 = (unsigned short*)alloc((size_t)HH * 4 * HH * 2);
    unsigned short* P_se  = (unsigned short*)alloc((size_t)32 * HH * 2);
    unsigned short* P_h1  = (unsigned short*)alloc((size_t)HH * PHH * 2);
    unsigned short* P_h2  = (unsigned short*)alloc((size_t)PHH * PHH * 2);
    unsigned short* P_log = (unsigned short*)alloc((size_t)PHH * 16 * 2);
    unsigned short* P_mu  = (unsigned short*)alloc((size_t)PHH * 48 * 2);
    unsigned short* P_sig = (unsigned short*)alloc((size_t)PHH * 48 * 2);
    unsigned short* P_pm  = (unsigned short*)alloc((size_t)HH * 16 * 2);
    unsigned short* P_ps  = (unsigned short*)alloc((size_t)HH * 16 * 2);

    auto pack = [&](const float* src, unsigned short* dst, int rK, int rN, int Kp, int Np,
                    int tr, float scale) {
        int total = Kp * Np;
        pack_b_kernel<<<(total + 255) / 256, 256, 0, stream>>>(src, dst, rK, rN, Kp, Np, tr, scale);
    };
    // gates = x @ W^T  -> B[k][n] = W[n][k]  (transposed=1)
    pack(Wih0, P_ih0, OBS, 4 * HH, OBS, 4 * HH, 1, 1.f);
    pack(Whh0, P_hh0, HH,  4 * HH, HH,  4 * HH, 1, 1.f);
    pack(Wih1, P_ih1, HH,  4 * HH, HH,  4 * HH, 1, 1.f);
    pack(Whh1, P_hh1, HH,  4 * HH, HH,  4 * HH, 1, 1.f);
    // y = x @ W (stored [K][N], transposed=0)
    pack(W_se, P_se, DD,  HH,  32,  HH,  0, 1.f);
    pack(Wh1,  P_h1, HH,  PHH, HH,  PHH, 0, 0.5f);   // fold infos 0.5
    pack(Wh2,  P_h2, PHH, PHH, PHH, PHH, 0, 1.f);
    pack(Wlog, P_log, PHH, KK, PHH, 16,  0, 1.f);
    pack(Wmu,  P_mu, PHH, 48, PHH, 48,  0, 1.f);
    pack(Wsig, P_sig, PHH, 48, PHH, 48, 0, 1.f);
    pack(W_pm, P_pm, HH, DD, HH, 16, 0, 0.5f);       // fold x0 = 0.5*hf[0]
    pack(W_ps, P_ps, HH, DD, HH, 16, 0, 0.5f);

    // ---- LSTM layer 0 (input = observations, f32) ----
    lstm_scan_kernel<<<8, 512, 0, stream>>>(obs, nullptr, P_ih0, OBS, P_hh0,
                                            b0, h0, c0, h1s, 1);
    // ---- LSTM layer 1 (input = h1s, bf16) ----
    lstm_scan_kernel<<<8, 512, 0, stream>>>(nullptr, h1s, P_ih1, HH, P_hh1,
                                            b1, h0 + HH, c0 + HH, hfb, 0);
    // ---- proposal scan ----
    proposal_scan_kernel<<<8, 256, 0, stream>>>(hfb, eps0, epsS, gumb,
                                                P_se, P_h1, P_h2, P_log, P_mu, P_sig,
                                                P_pm, P_ps,
                                                b_se, bh1, bh2, blog, bmu, bsig, b_pm, b_ps,
                                                out);
}
